// SAGE_43078521979009
// MI455X (gfx1250) — compile-verified, matrix operand order checked by reading, not run on hardware
//
#include <hip/hip_runtime.h>
#include <hip/hip_bf16.h>

typedef __attribute__((ext_vector_type(16))) _Float16 v16h;
typedef __attribute__((ext_vector_type(4)))  _Float16 v4h;
typedef __attribute__((ext_vector_type(8)))  float    v8f;

#define N1_SRC 495616
#define N1_DST 45056
#define E1     450560
#define N2_DST 4096
#define E2     40960
#define D_IN   512
#define D_HID  512
#define D_OUT  256

// ---------------------------------------------------------------------------
// Zero-fill (grid-stride, 32-bit words)
// ---------------------------------------------------------------------------
__global__ void zero_words(unsigned int* __restrict__ p, size_t nwords) {
    size_t i = (size_t)blockIdx.x * blockDim.x + threadIdx.x;
    size_t stride = (size_t)gridDim.x * blockDim.x;
    for (; i < nwords; i += stride) p[i] = 0u;
}

// ---------------------------------------------------------------------------
// Edge scatter-add: msg[dst] += feats[src]; deg[dst] += 1 (once per edge).
// One block per edge; 128 threads * float4 = 512 floats. Row reads are
// contiguous 2KB (coalesced); atomics land in L2 (msg fits in 192MB L2).
// ---------------------------------------------------------------------------
__global__ __launch_bounds__(128)
void scatter_add(const float* __restrict__ src_feats,
                 const int* __restrict__ esrc,
                 const int* __restrict__ edst,
                 float* __restrict__ msg,
                 int* __restrict__ deg,
                 int n_edges) {
    int e = blockIdx.x;
    if (e >= n_edges) return;
    int s = esrc[e];
    int d = edst[e];
    const float4* __restrict__ srow = (const float4*)(src_feats + (size_t)s * 512);
    float* __restrict__ drow = msg + (size_t)d * 512;
    float4 v = srow[threadIdx.x];
    int c = threadIdx.x * 4;
    atomicAdd(&drow[c + 0], v.x);
    atomicAdd(&drow[c + 1], v.y);
    atomicAdd(&drow[c + 2], v.z);
    atomicAdd(&drow[c + 3], v.w);
    if (threadIdx.x == 0) atomicAdd(&deg[d], 1);
}

// ---------------------------------------------------------------------------
// agg = (msg + h_dst) / (deg + 1), converted to f16 (WMMA A-matrix feed)
// ---------------------------------------------------------------------------
__global__ __launch_bounds__(128)
void agg_to_f16(const float* __restrict__ msg,
                const float* __restrict__ hdst,
                const int* __restrict__ deg,
                _Float16* __restrict__ out,
                int n_rows) {
    int r = blockIdx.x;
    if (r >= n_rows) return;
    float inv = 1.0f / ((float)deg[r] + 1.0f);
    const float4* __restrict__ m = (const float4*)(msg  + (size_t)r * 512);
    const float4* __restrict__ h = (const float4*)(hdst + (size_t)r * 512);
    v4h* __restrict__ o = (v4h*)(out + (size_t)r * 512);
    float4 a = m[threadIdx.x];
    float4 b = h[threadIdx.x];
    v4h q;
    q[0] = (_Float16)((a.x + b.x) * inv);
    q[1] = (_Float16)((a.y + b.y) * inv);
    q[2] = (_Float16)((a.z + b.z) * inv);
    q[3] = (_Float16)((a.w + b.w) * inv);
    o[threadIdx.x] = q;
}

// ---------------------------------------------------------------------------
// Weight convert + transpose: WT[n][k] = (f16) W[k][n].
// W is tiny (<=1MB) and read once per call; simple per-element kernel.
// ---------------------------------------------------------------------------
__global__ void w_to_f16_T(const float* __restrict__ W, _Float16* __restrict__ WT,
                           int K, int N) {
    int i = blockIdx.x * blockDim.x + threadIdx.x;   // i = k*N + n (coalesced read)
    if (i < K * N) {
        int k = i / N, n = i % N;
        WT[(size_t)n * K + k] = (_Float16)W[i];
    }
}

// ---------------------------------------------------------------------------
// WMMA GEMM: C = act(A(MxK,f16) * B(KxN,f16 given as BT = NxK) + bias) -> f32
// Block: 128 threads (4 waves), 64x64 tile, K-step 32.
// Both A and BT are staged K-contiguous, so A and B fragments each lower to
// two ds_load_b128 (CDNA5 16-bit A 16x32 / B 32x16 per-lane layouts are
// symmetric: lanes 0-15 hold K {0..7,16..23}, lanes 16-31 hold {8..15,24..31}).
// M%64==0, N%64==0, K%32==0 guaranteed by caller -> no divergence (EXEC all 1s).
// ---------------------------------------------------------------------------
#define TM 64
#define TN 64
#define TK 32
#define LDP 40   // padded LDS row pitch in halves (80B, 16B-aligned, spreads banks)

__global__ __launch_bounds__(128)
void gemm_bias_act(const _Float16* __restrict__ A,    // M x K row-major
                   const _Float16* __restrict__ BT,   // N x K row-major (B transposed)
                   const float* __restrict__ bias,    // N
                   float* __restrict__ C,             // M x N row-major
                   int M, int N, int K, int do_relu) {
    __shared__ _Float16 As[TM][LDP];
    __shared__ _Float16 Bs[TN][LDP];

    const int tid  = threadIdx.x;
    const int wave = tid >> 5;            // 0..3
    const int lane = tid & 31;
    const int m0 = blockIdx.y * TM;
    const int n0 = blockIdx.x * TN;

    v8f acc0 = {}, acc1 = {}, acc2 = {}, acc3 = {};

    const int idx16 = lane & 15;          // M (A frag) / N (B frag) within 16
    const int kbase = (lane >> 4) << 3;   // 0 or 8

    // Staging coordinates: each thread copies 16 contiguous halves (2x uint4).
    const int sr = tid >> 1;              // 0..63
    const int sc = (tid & 1) * 16;        // 0 or 16

    for (int k0 = 0; k0 < K; k0 += TK) {
        {
            const uint4* __restrict__ src = (const uint4*)(A + (size_t)(m0 + sr) * K + k0 + sc);
            *(uint4*)&As[sr][sc + 0] = src[0];
            *(uint4*)&As[sr][sc + 8] = src[1];
        }
        {
            const uint4* __restrict__ src = (const uint4*)(BT + (size_t)(n0 + sr) * K + k0 + sc);
            *(uint4*)&Bs[sr][sc + 0] = src[0];
            *(uint4*)&Bs[sr][sc + 8] = src[1];
        }
        __syncthreads();

        // A fragment: two contiguous 8-half runs at k = kbase and kbase+16.
        v16h afrag;
        {
            const _Float16* __restrict__ ap = &As[wave * 16 + idx16][0];
            #pragma unroll
            for (int i = 0; i < 8; ++i) afrag[i]     = ap[kbase + i];
            #pragma unroll
            for (int i = 0; i < 8; ++i) afrag[8 + i] = ap[kbase + 16 + i];
        }
        #pragma unroll
        for (int nt = 0; nt < 4; ++nt) {
            v16h bfrag;
            const _Float16* __restrict__ bp = &Bs[nt * 16 + idx16][0];
            #pragma unroll
            for (int i = 0; i < 8; ++i) bfrag[i]     = bp[kbase + i];
            #pragma unroll
            for (int i = 0; i < 8; ++i) bfrag[8 + i] = bp[kbase + 16 + i];
            if (nt == 0) acc0 = __builtin_amdgcn_wmma_f32_16x16x32_f16(false, afrag, false, bfrag, (short)0, acc0, false, false);
            if (nt == 1) acc1 = __builtin_amdgcn_wmma_f32_16x16x32_f16(false, afrag, false, bfrag, (short)0, acc1, false, false);
            if (nt == 2) acc2 = __builtin_amdgcn_wmma_f32_16x16x32_f16(false, afrag, false, bfrag, (short)0, acc2, false, false);
            if (nt == 3) acc3 = __builtin_amdgcn_wmma_f32_16x16x32_f16(false, afrag, false, bfrag, (short)0, acc3, false, false);
        }
        __syncthreads();
    }

    // Write-out: f32 C/D layout -> VGPR v: lanes 0-15 row=v, lanes 16-31 row=v+8.
    const int rbase = m0 + wave * 16 + ((lane >> 4) << 3);
    #pragma unroll
    for (int nt = 0; nt < 4; ++nt) {
        v8f acc = (nt == 0) ? acc0 : (nt == 1) ? acc1 : (nt == 2) ? acc2 : acc3;
        int col = n0 + nt * 16 + idx16;
        float bv = bias[col];
        #pragma unroll
        for (int v = 0; v < 8; ++v) {
            float x = acc[v] + bv;
            if (do_relu) x = fmaxf(x, 0.0f);
            C[(size_t)(rbase + v) * N + col] = x;
        }
    }
}

// ---------------------------------------------------------------------------
// Host-side orchestration
// ---------------------------------------------------------------------------
static inline size_t align256(size_t x) { return (x + 255) & ~(size_t)255; }

extern "C" void kernel_launch(void* const* d_in, const int* in_sizes, int n_in,
                              void* d_out, int out_size, void* d_ws, size_t ws_size,
                              hipStream_t stream) {
    (void)in_sizes; (void)n_in; (void)out_size; (void)ws_size;

    const float* feats = (const float*)d_in[0];   // N1_SRC x D_IN
    const float* W1    = (const float*)d_in[1];   // D_IN x D_HID
    const float* b1    = (const float*)d_in[2];   // D_HID
    const float* W2    = (const float*)d_in[3];   // D_HID x D_OUT
    const float* b2    = (const float*)d_in[4];   // D_OUT
    const int* esrc1   = (const int*)d_in[5];
    const int* edst1   = (const int*)d_in[6];
    const int* esrc2   = (const int*)d_in[7];
    const int* edst2   = (const int*)d_in[8];
    float* out = (float*)d_out;                   // N2_DST x D_OUT

    // Workspace carve-up (msg1/deg1/msg2/deg2 contiguous for one zero pass).
    char* ws = (char*)d_ws;
    size_t off = 0;
    float*    msg1 = (float*)(ws + off); off += (size_t)N1_DST * D_IN  * sizeof(float);
    int*      deg1 = (int*)  (ws + off); off += (size_t)N1_DST * sizeof(int);
    float*    msg2 = (float*)(ws + off); off += (size_t)N2_DST * D_HID * sizeof(float);
    int*      deg2 = (int*)  (ws + off);
    size_t zero_n = (off + (size_t)N2_DST * sizeof(int)) / 4;
    off = align256(off + (size_t)N2_DST * sizeof(int));
    _Float16* agg1 = (_Float16*)(ws + off); off += align256((size_t)N1_DST * D_IN  * sizeof(_Float16));
    _Float16* agg2 = (_Float16*)(ws + off); off += align256((size_t)N2_DST * D_HID * sizeof(_Float16));
    _Float16* W1T  = (_Float16*)(ws + off); off += align256((size_t)D_IN  * D_HID * sizeof(_Float16));
    _Float16* W2T  = (_Float16*)(ws + off); off += align256((size_t)D_HID * D_OUT * sizeof(_Float16));
    float*    h    = (float*)   (ws + off); off += align256((size_t)N1_DST * D_HID * sizeof(float));

    // 1) Zero accumulators (fresh every call; harness does not re-zero ws).
    zero_words<<<2048, 256, 0, stream>>>((unsigned int*)msg1, zero_n);

    // 2) Weight convert+transpose (independent of scatter).
    w_to_f16_T<<<(D_IN * D_HID + 255) / 256, 256, 0, stream>>>(W1, W1T, D_IN, D_HID);
    w_to_f16_T<<<(D_HID * D_OUT + 255) / 256, 256, 0, stream>>>(W2, W2T, D_HID, D_OUT);

    // 3) Layer 1: scatter, aggregate, GEMM (+bias, ReLU) -> h
    scatter_add<<<E1, 128, 0, stream>>>(feats, esrc1, edst1, msg1, deg1, E1);
    agg_to_f16<<<N1_DST, 128, 0, stream>>>(msg1, feats, deg1, agg1, N1_DST);
    {
        dim3 grid(D_HID / TN, N1_DST / TM);
        gemm_bias_act<<<grid, 128, 0, stream>>>(agg1, W1T, b1, h, N1_DST, D_HID, D_IN, 1);
    }

    // 4) Layer 2: scatter, aggregate, GEMM (+bias) -> out
    scatter_add<<<E2, 128, 0, stream>>>(h, esrc2, edst2, msg2, deg2, E2);
    agg_to_f16<<<N2_DST, 128, 0, stream>>>(msg2, h, deg2, agg2, N2_DST);
    {
        dim3 grid(D_OUT / TN, N2_DST / TM);
        gemm_bias_act<<<grid, 128, 0, stream>>>(agg2, W2T, b2, out, N2_DST, D_OUT, D_IN, 0);
    }
}
